// StructureEncoder_23545010717384
// MI455X (gfx1250) — compile-verified, matrix operand order checked by reading, not run on hardware
//
#include <hip/hip_runtime.h>
#include <hip/hip_bf16.h>
#include <cstdint>

typedef unsigned short u16;
typedef unsigned int   u32;
typedef unsigned long long u64;
typedef long long      LL;

typedef __attribute__((ext_vector_type(16))) __bf16 v16bf;
typedef __attribute__((ext_vector_type(8)))  float  v8f;
typedef __attribute__((ext_vector_type(4)))  unsigned int v4u;
typedef __attribute__((ext_vector_type(8)))  int v8i;
typedef __attribute__((ext_vector_type(4)))  int v4i;

#if defined(__HIP_DEVICE_COMPILE__) && __has_builtin(__builtin_amdgcn_tensor_load_to_lds) && __has_builtin(__builtin_amdgcn_s_wait_tensorcnt)
#define GEMM_TDM 1
#else
#define GEMM_TDM 0
#endif

// ---------------- bf16 helpers (RNE) ----------------
__device__ __forceinline__ u16 f2bf(float f) {
  u32 u = __builtin_bit_cast(u32, f);
  return (u16)((u + 0x7fffu + ((u >> 16) & 1u)) >> 16);
}
__device__ __forceinline__ float bf2f(u16 h) {
  u32 u = ((u32)h) << 16;
  return __builtin_bit_cast(float, u);
}

// =====================================================================
// WMMA GEMM:  C = alpha * A(MxK) * B(NxK)^T + bias,  optional relu,
// optional bf16 store. Two-level batching: z -> (z/binner, z%binner).
// A,B bf16 row-major (lda/ldb elements, even). Tiles staged into LDS by
// the Tensor Data Mover (D# descriptor, TENSORcnt) with double buffering;
// TDM zero-fills M/N/K tails via tensor_dim OOB semantics.
// =====================================================================
#define BM 128
#define BN 64
#define BKK 64

struct GemmP {
  const u16* A; const u16* B; const float* bias; void* C;
  int M, N, K, lda, ldb, ldc;
  LL sA1, sA2, sB1, sB2, sC1, sC2;
  int binner; float alpha; int relu; int c_bf16;
};

#if GEMM_TDM
// Issue one 2D bf16 tile DMA: global (tensor_d1 x tensor_d0, stride) -> LDS
// packed tile (tile_d1 x tile_d0). OOB reads return zero (tail handling).
__device__ __forceinline__ void tdm_load_2d(const u16* gptr, u32 lds_off,
                                            u32 tensor_d0, u32 tensor_d1,
                                            u64 stride_elems, u32 tile_d0, u32 tile_d1) {
  u64 ga = (u64)(size_t)gptr;
  v4u g0;
  g0[0] = 1u;                                        // count=1, user mode, no gather
  g0[1] = lds_off;                                   // lds_addr (bytes)
  g0[2] = (u32)ga;                                   // global_addr[31:0]
  g0[3] = (u32)((ga >> 32) & 0x01FFFFFFu) | (2u << 30); // global_addr[56:32] | type=2
  v8i g1;
  g1[0] = (int)(1u << 16);                           // workgroup_mask=0, data_size=1 (2B)
  g1[1] = (int)((tensor_d0 & 0xFFFFu) << 16);        // tensor_dim0[15:0] @ bits 63:48
  g1[2] = (int)(((tensor_d0 >> 16) & 0xFFFFu) | ((tensor_d1 & 0xFFFFu) << 16));
  g1[3] = (int)(((tensor_d1 >> 16) & 0xFFFFu) | ((tile_d0 & 0xFFFFu) << 16));
  g1[4] = (int)(tile_d1 & 0xFFFFu);                  // tile_dim1; tile_dim2=0
  g1[5] = (int)(u32)(stride_elems & 0xFFFFFFFFull);  // dim0_stride[31:0]
  g1[6] = (int)(u32)((stride_elems >> 32) & 0xFFFFull); // dim0_stride[47:32]; dim1_stride lo=0
  g1[7] = 0;
  v4i zz = {0, 0, 0, 0};
  v8i z8 = {0, 0, 0, 0, 0, 0, 0, 0};
  __builtin_amdgcn_tensor_load_to_lds(g0, g1, zz, zz, z8, 0);
}
#endif

// Load one 16x32 bf16 fragment (rows x K) from LDS tile (row-major, ld even).
// ISA layout: lane r(0..15) -> row M=r, K 0..7/16..23; lanes 16..31 -> K+8.
__device__ __forceinline__ v16bf load_frag(const u16* p, int ld, int lane) {
  int r   = lane & 15;
  int hi8 = ((lane >> 4) & 1) * 8;
  union { u32 u[8]; v16bf v; } f;
  const u16* row = p + r * ld;
#pragma unroll
  for (int j = 0; j < 8; ++j) {
    int k = ((j >> 2) << 4) + hi8 + ((j & 3) << 1);
    f.u[j] = *(const u32*)(row + k);
  }
  return f.v;
}

__global__ __launch_bounds__(256) void k_gemm_nt(GemmP p) {
  __shared__ __align__(16) u16 As[2][BM * BKK];
  __shared__ __align__(16) u16 Bs[2][BN * BKK];
  int bz = blockIdx.z;
  int bo = bz / p.binner, bi = bz % p.binner;
  const u16* A = p.A + bo * p.sA1 + bi * p.sA2;
  const u16* B = p.B + bo * p.sB1 + bi * p.sB2;
  LL cbase = bo * p.sC1 + bi * p.sC2;
  int m0 = blockIdx.y * BM, n0 = blockIdx.x * BN;
  int tid = threadIdx.x, lane = tid & 31, wave = tid >> 5;
  int wm = wave >> 1, wn = wave & 1;                 // 4x2 waves, 32x32 per wave
  v8f zero = {0.f,0.f,0.f,0.f,0.f,0.f,0.f,0.f};
  v8f acc[2][2];
#pragma unroll
  for (int i = 0; i < 2; ++i)
#pragma unroll
    for (int j = 0; j < 2; ++j) acc[i][j] = zero;

  int nk = (p.K + BKK - 1) / BKK;

#if GEMM_TDM
  u32 ldsA[2] = { (u32)(size_t)&As[0][0], (u32)(size_t)&As[1][0] };
  u32 ldsB[2] = { (u32)(size_t)&Bs[0][0], (u32)(size_t)&Bs[1][0] };
  if (wave == 0) {
    tdm_load_2d(A + (LL)m0 * p.lda, ldsA[0], (u32)p.K, (u32)(p.M - m0),
                (u64)p.lda, BKK, BM);
    tdm_load_2d(B + (LL)n0 * p.ldb, ldsB[0], (u32)p.K, (u32)(p.N - n0),
                (u64)p.ldb, BKK, BN);
  }
  for (int t = 0; t < nk; ++t) {
    int pb = t & 1;
    if (wave == 0) __builtin_amdgcn_s_wait_tensorcnt(0);
    __syncthreads();
    if (wave == 0 && t + 1 < nk) {
      int k1 = (t + 1) * BKK;
      tdm_load_2d(A + (LL)m0 * p.lda + k1, ldsA[pb ^ 1], (u32)(p.K - k1),
                  (u32)(p.M - m0), (u64)p.lda, BKK, BM);
      tdm_load_2d(B + (LL)n0 * p.ldb + k1, ldsB[pb ^ 1], (u32)(p.K - k1),
                  (u32)(p.N - n0), (u64)p.ldb, BKK, BN);
    }
#pragma unroll
    for (int kk = 0; kk < BKK; kk += 32) {
      v16bf af0 = load_frag(&As[pb][(wm * 32 +  0) * BKK + kk], BKK, lane);
      v16bf af1 = load_frag(&As[pb][(wm * 32 + 16) * BKK + kk], BKK, lane);
      v16bf bf0 = load_frag(&Bs[pb][(wn * 32 +  0) * BKK + kk], BKK, lane);
      v16bf bf1 = load_frag(&Bs[pb][(wn * 32 + 16) * BKK + kk], BKK, lane);
      acc[0][0] = __builtin_amdgcn_wmma_f32_16x16x32_bf16(false, af0, false, bf0, (short)0, acc[0][0], false, false);
      acc[0][1] = __builtin_amdgcn_wmma_f32_16x16x32_bf16(false, af0, false, bf1, (short)0, acc[0][1], false, false);
      acc[1][0] = __builtin_amdgcn_wmma_f32_16x16x32_bf16(false, af1, false, bf0, (short)0, acc[1][0], false, false);
      acc[1][1] = __builtin_amdgcn_wmma_f32_16x16x32_bf16(false, af1, false, bf1, (short)0, acc[1][1], false, false);
    }
    __syncthreads();
  }
#else
  // Fallback: cooperative staging (no TDM builtin available).
  for (int t = 0; t < nk; ++t) {
    int pb = t & 1;
    int k0 = t * BKK;
#pragma unroll
    for (int it = 0; it < (BM * BKK / 2) / 256; ++it) {
      int idx = tid + it * 256;
      int r = idx / (BKK / 2), kc = (idx % (BKK / 2)) * 2;
      int gm = m0 + r, gk = k0 + kc;
      u32 val = 0;
      if (gm < p.M && gk + 1 < p.K) val = *(const u32*)(A + (LL)gm * p.lda + gk);
      else if (gm < p.M && gk < p.K) val = (u32)A[(LL)gm * p.lda + gk];
      *(u32*)&As[pb][r * BKK + kc] = val;
    }
#pragma unroll
    for (int it = 0; it < (BN * BKK / 2) / 256; ++it) {
      int idx = tid + it * 256;
      int r = idx / (BKK / 2), kc = (idx % (BKK / 2)) * 2;
      int gn = n0 + r, gk = k0 + kc;
      u32 val = 0;
      if (gn < p.N && gk + 1 < p.K) val = *(const u32*)(B + (LL)gn * p.ldb + gk);
      else if (gn < p.N && gk < p.K) val = (u32)B[(LL)gn * p.ldb + gk];
      *(u32*)&Bs[pb][r * BKK + kc] = val;
    }
    __syncthreads();
#pragma unroll
    for (int kk = 0; kk < BKK; kk += 32) {
      v16bf af0 = load_frag(&As[pb][(wm * 32 +  0) * BKK + kk], BKK, lane);
      v16bf af1 = load_frag(&As[pb][(wm * 32 + 16) * BKK + kk], BKK, lane);
      v16bf bf0 = load_frag(&Bs[pb][(wn * 32 +  0) * BKK + kk], BKK, lane);
      v16bf bf1 = load_frag(&Bs[pb][(wn * 32 + 16) * BKK + kk], BKK, lane);
      acc[0][0] = __builtin_amdgcn_wmma_f32_16x16x32_bf16(false, af0, false, bf0, (short)0, acc[0][0], false, false);
      acc[0][1] = __builtin_amdgcn_wmma_f32_16x16x32_bf16(false, af0, false, bf1, (short)0, acc[0][1], false, false);
      acc[1][0] = __builtin_amdgcn_wmma_f32_16x16x32_bf16(false, af1, false, bf0, (short)0, acc[1][0], false, false);
      acc[1][1] = __builtin_amdgcn_wmma_f32_16x16x32_bf16(false, af1, false, bf1, (short)0, acc[1][1], false, false);
    }
    __syncthreads();
  }
#endif

  // epilogue: C/D layout lane = n + 16*(m>>3), vgpr j holds m = (lane>>4)*8 + j
  int nl = lane & 15, mseg = (lane >> 4) * 8;
#pragma unroll
  for (int i = 0; i < 2; ++i) {
    int mbase = m0 + wm * 32 + i * 16 + mseg;
#pragma unroll
    for (int j = 0; j < 2; ++j) {
      int gn = n0 + wn * 32 + j * 16 + nl;
      if (gn >= p.N) continue;
      float bv = p.bias ? p.bias[gn] : 0.f;
#pragma unroll
      for (int r = 0; r < 8; ++r) {
        int gm = mbase + r;
        if (gm >= p.M) continue;
        float v = acc[i][j][r] * p.alpha + bv;
        if (p.relu) v = v > 0.f ? v : 0.f;
        LL off = cbase + (LL)gm * p.ldc + gn;
        if (p.c_bf16) ((u16*)p.C)[off] = f2bf(v);
        else          ((float*)p.C)[off] = v;
      }
    }
  }
}

// ---------------- support kernels ----------------
__global__ void k_cvt_bf16(const float* __restrict__ in, u16* __restrict__ out, LL n) {
  for (LL e = blockIdx.x * (LL)blockDim.x + threadIdx.x; e < n; e += (LL)gridDim.x * blockDim.x)
    out[e] = f2bf(in[e]);
}

// out[b][c*R + r] = cvt(in[(b/binner)*ibs1 + (b%binner)*ibs2 + r*irs + c*ics])
__global__ void k_transpose_any(const void* in, u16* out, int R, int Cc,
                                LL irs, LL ics, LL ibs1, LL ibs2, int binner, LL obs, int in_bf16) {
  LL b = blockIdx.y, bo = b / binner, bi = b % binner;
  LL src = bo * ibs1 + bi * ibs2;
  u16* dst = out + b * obs;
  LL total = (LL)R * Cc;
  for (LL i = blockIdx.x * (LL)blockDim.x + threadIdx.x; i < total; i += (LL)gridDim.x * blockDim.x) {
    int r = (int)(i / Cc), c = (int)(i % Cc);
    LL si = src + (LL)r * irs + (LL)c * ics;
    float v = in_bf16 ? bf2f(((const u16*)in)[si]) : ((const float*)in)[si];
    dst[(LL)c * R + r] = f2bf(v);
  }
}

// y = LN(x + resid) * optional mask; one block per row
__global__ void k_layernorm(const float* __restrict__ x, const float* __restrict__ resid,
                            const float* __restrict__ g, const float* __restrict__ bt,
                            float* __restrict__ y, int D, const float* rm, int N, int mask2d) {
  LL row = blockIdx.x; int tid = threadIdx.x;
  float v = 0.f;
  if (tid < D) { v = x[row * D + tid]; if (resid) v += resid[row * D + tid]; }
  __shared__ float red[256];
  red[tid] = (tid < D) ? v : 0.f; __syncthreads();
  for (int s = 128; s > 0; s >>= 1) { if (tid < s) red[tid] += red[tid + s]; __syncthreads(); }
  float mean = red[0] / D; __syncthreads();
  float d = (tid < D) ? (v - mean) : 0.f;
  red[tid] = d * d; __syncthreads();
  for (int s = 128; s > 0; s >>= 1) { if (tid < s) red[tid] += red[tid + s]; __syncthreads(); }
  float inv = rsqrtf(red[0] / D + 1e-5f);
  if (tid < D) {
    float o = (v - mean) * inv * g[tid] + bt[tid];
    if (rm) o *= mask2d ? rm[(row / N) % N] * rm[row % N] : rm[row % N];
    y[row * D + tid] = o;
  }
}

__global__ void k_softmax_bf16(const float* __restrict__ x, const float* addv,
                               u16* __restrict__ out, int N) {
  LL row = blockIdx.x; int tid = threadIdx.x;
  float v = -3.0e38f;
  if (tid < N) v = x[row * N + tid] + (addv ? addv[tid] : 0.f);
  __shared__ float red[256];
  red[tid] = v; __syncthreads();
  for (int s = 128; s > 0; s >>= 1) { if (tid < s) red[tid] = fmaxf(red[tid], red[tid + s]); __syncthreads(); }
  float mx = red[0]; __syncthreads();
  float e = (tid < N) ? __expf(v - mx) : 0.f;
  red[tid] = e; __syncthreads();
  for (int s = 128; s > 0; s >>= 1) { if (tid < s) red[tid] += red[tid + s]; __syncthreads(); }
  float inv = 1.f / red[0];
  if (tid < N) out[row * N + tid] = f2bf(e * inv);
}

__global__ void k_index_embed(float* out, LL total, int d, int mode, int N) {
  for (LL e = blockIdx.x * (LL)blockDim.x + threadIdx.x; e < total; e += (LL)gridDim.x * blockDim.x) {
    int c = (int)(e % d); LL row = e / d;
    float idx;
    if (mode == 0) idx = (float)row;
    else { int i = (int)(row / N), j = (int)(row % N); idx = (float)(i - j); }
    int half = d >> 1; int k = c % half;
    float ang = idx * 3.14159265358979f / __powf(2056.f, 2.f * (float)k / (float)d);
    out[e] = (c < half) ? __sinf(ang) : __cosf(ang);
  }
}

// edge-proj input features: [p_i[i](64), p_i[j](64), relpos[i,j](64), distogram(22)] -> bf16
__global__ void k_edge_feats(const float* __restrict__ p_i, const float* __restrict__ relpos,
                             const float* __restrict__ trans, u16* __restrict__ out, int BKc, int N) {
  LL total = (LL)BKc * N * N * 214;
  for (LL e = blockIdx.x * (LL)blockDim.x + threadIdx.x; e < total; e += (LL)gridDim.x * blockDim.x) {
    int c = (int)(e % 214); LL r = e / 214;
    int j = (int)(r % N); r /= N; int i = (int)(r % N); int b = (int)(r / N);
    float v;
    if (c < 64)       v = p_i[i * 64 + c];
    else if (c < 128) v = p_i[j * 64 + (c - 64)];
    else if (c < 192) v = relpos[((LL)i * N + j) * 64 + (c - 128)];
    else {
      int m = c - 192;
      const float* ti = trans + ((LL)b * N + i) * 3;
      const float* tj = trans + ((LL)b * N + j) * 3;
      float dx = ti[0]-tj[0], dy = ti[1]-tj[1], dz = ti[2]-tj[2];
      float d = sqrtf(dx*dx + dy*dy + dz*dz);
      float step = 19.999f / 21.f;
      float lo = 0.001f + m * step;
      float up = (m == 21) ? 1e8f : 0.001f + (m + 1) * step;
      v = (d > lo && d < up) ? 1.f : 0.f;
    }
    out[e] = f2bf(v);
  }
}

// out[tk,p,:] = R[tk] @ (x,y,z) + 0.1*trans[tk] ; raw (tk, 3P) split x|y|z
__global__ void k_rot_points(const float* __restrict__ raw, const float* __restrict__ R,
                             const float* __restrict__ trans, float* __restrict__ out, LL tokens, int P) {
  LL total = tokens * P;
  for (LL e = blockIdx.x * (LL)blockDim.x + threadIdx.x; e < total; e += (LL)gridDim.x * blockDim.x) {
    int p = (int)(e % P); LL tk = e / P;
    float x = raw[tk * 3 * P + p], y = raw[tk * 3 * P + P + p], z = raw[tk * 3 * P + 2 * P + p];
    const float* Rm = R + tk * 9;
    float t0 = trans[tk*3]*0.1f, t1 = trans[tk*3+1]*0.1f, t2 = trans[tk*3+2]*0.1f;
    out[e*3+0] = Rm[0]*x + Rm[1]*y + Rm[2]*z + t0;
    out[e*3+1] = Rm[3]*x + Rm[4]*y + Rm[5]*z + t1;
    out[e*3+2] = Rm[6]*x + Rm[7]*y + Rm[8]*z + t2;
  }
}

// logits += sqrt(1/3)*b_proj + point-distance bias + INF*(m_i*m_j - 1)
__global__ void k_pt_bias(float* __restrict__ logits, const float* __restrict__ bproj,
                          const float* __restrict__ qp, const float* __restrict__ kvp,
                          const float* __restrict__ hw, const float* __restrict__ rm, int BKc, int N) {
  LL total = (LL)BKc * 8 * N * N;
  for (LL e = blockIdx.x * (LL)blockDim.x + threadIdx.x; e < total; e += (LL)gridDim.x * blockDim.x) {
    int j = (int)(e % N); LL r = e / N;
    int i = (int)(r % N); r /= N; int h = (int)(r % 8); int b = (int)(r / 8);
    float sp = logf(1.f + __expf(hw[h]));
    const float* qpp = qp  + ((LL)(b * N + i) * 64  + h * 8)  * 3;
    const float* kpp = kvp + ((LL)(b * N + j) * 160 + h * 20) * 3;
    float s = 0.f;
#pragma unroll
    for (int p = 0; p < 8; ++p) {
      float dx = qpp[p*3]-kpp[p*3], dy = qpp[p*3+1]-kpp[p*3+1], dz = qpp[p*3+2]-kpp[p*3+2];
      s += dx*dx + dy*dy + dz*dz;
    }
    float pt   = -0.5f * sp * 0.09622504486493764f * s;       // sqrt(1/108)
    float bias = 100000.f * (rm[i] * rm[j] - 1.f);
    logits[e] += 0.5773502691896258f * bproj[((LL)(b * N + i) * N + j) * 8 + h] + pt + bias;
  }
}

// concat [o(1024), opx(96), opy(96), opz(96), opnorm(96), opair(1024)] with inverse-frame rotation
__global__ void k_gather_feats(const float* __restrict__ o, const float* __restrict__ op,
                               const float* __restrict__ opair, const float* __restrict__ R,
                               const float* __restrict__ trans, u16* __restrict__ out, LL tokens) {
  LL total = tokens * 2432;
  for (LL e = blockIdx.x * (LL)blockDim.x + threadIdx.x; e < total; e += (LL)gridDim.x * blockDim.x) {
    int c = (int)(e % 2432); LL tk = e / 2432;
    float v;
    if (c < 1024) v = o[tk * 1024 + c];
    else if (c < 1408) {
      int cc = c - 1024; int comp = cc / 96; int hp = cc % 96;
      const float* Rm = R + tk * 9;
      float tx = trans[tk*3]*0.1f, ty = trans[tk*3+1]*0.1f, tz = trans[tk*3+2]*0.1f;
      const float* opp = op + tk * 288 + hp * 3;
      float vx = opp[0]-tx, vy = opp[1]-ty, vz = opp[2]-tz;
      float rx = Rm[0]*vx + Rm[3]*vy + Rm[6]*vz;   // R^T v
      float ry = Rm[1]*vx + Rm[4]*vy + Rm[7]*vz;
      float rz = Rm[2]*vx + Rm[5]*vy + Rm[8]*vz;
      if (comp == 0) v = rx; else if (comp == 1) v = ry; else if (comp == 2) v = rz;
      else v = sqrtf(rx*rx + ry*ry + rz*rz + 1e-8f);
    } else v = opair[tk * 1024 + (c - 1408)];
    out[e] = f2bf(v);
  }
}

__global__ void k_add(float* out, const float* a, const float* b, LL n) {
  for (LL e = blockIdx.x * (LL)blockDim.x + threadIdx.x; e < n; e += (LL)gridDim.x * blockDim.x)
    out[e] = a[e] + b[e];
}
__global__ void k_maskrow(float* x, const float* rm, LL total, int D, int N) {
  for (LL e = blockIdx.x * (LL)blockDim.x + threadIdx.x; e < total; e += (LL)gridDim.x * blockDim.x)
    x[e] *= rm[(e / D) % N];
}
__global__ void k_bcast_mask(const float* src, const float* rm, float* dst, int copies, int Nrow, int D) {
  LL total = (LL)copies * Nrow * D;
  for (LL e = blockIdx.x * (LL)blockDim.x + threadIdx.x; e < total; e += (LL)gridDim.x * blockDim.x) {
    int c = (int)(e % D); LL row = e / D; int i = (int)(row % Nrow);
    dst[e] = src[(LL)i * D + c] * rm[i];
  }
}
__global__ void k_maskbias(const float* rm, float* mb, int N, float negval) {
  int i = threadIdx.x;
  if (i < N) mb[i] = rm[i] > 0.5f ? 0.f : negval;
}
// ee = [edge(128), nb_i(128), nb_j(128)] (+ optional bf16 addend) -> bf16
__global__ void k_edge_concat(const float* __restrict__ edge, const float* __restrict__ nb,
                              const u16* add, u16* __restrict__ out, int BKc, int N) {
  LL total = (LL)BKc * N * N * 384;
  for (LL e = blockIdx.x * (LL)blockDim.x + threadIdx.x; e < total; e += (LL)gridDim.x * blockDim.x) {
    int c = (int)(e % 384); LL r = e / 384;
    int j = (int)(r % N); r /= N; int i = (int)(r % N); int b = (int)(r / N);
    float v;
    if (c < 128)      v = edge[(((LL)b * N + i) * N + j) * 128 + c];
    else if (c < 256) v = nb[((LL)b * N + i) * 128 + (c - 128)];
    else              v = nb[((LL)b * N + j) * 128 + (c - 256)];
    if (add) v += bf2f(add[e]);
    out[e] = f2bf(v);
  }
}

// ---------------- host orchestration ----------------
namespace {
constexpr int BKC = 4, NSEQ = 256, CS = 256, CZ = 128, FEATC = 64;
constexpr int HHn = 8, CHn = 128, CPOSC = 128;
constexpr LL NNe = (LL)NSEQ * NSEQ;       // 65536
constexpr LL BKNN = (LL)BKC * NNe;        // 262144
constexpr LL BKN = (LL)BKC * NSEQ;        // 1024

struct Lin { const float* w; const float* b; };
struct LNp { const float* g; const float* b; };
struct Tf  { const float* b_in; const float* b_out; Lin l1, l2; LNp ln1, ln2; const float* w_in; const float* w_out; };
struct Blk {
  Lin et_final, et_init, et_t1, et_t2; LNp et_ln; int has_et;
  Lin ipa_b; const float* ipa_hw; Lin ipa_kv, ipa_kvp, ipa_out, ipa_q, ipa_qp;
  LNp ipa_ln; Lin nt1, nt2, nt3; LNp nt_ln; Lin post; Tf tf[2];
};
} // namespace

extern "C" void kernel_launch(void* const* d_in, const int* in_sizes, int n_in,
                              void* d_out, int out_size, void* d_ws, size_t ws_size,
                              hipStream_t stream) {
  (void)in_sizes; (void)n_in; (void)out_size; (void)ws_size;
  const float* trans = (const float*)d_in[0];   // (BK*N, 3)
  const float* rot   = (const float*)d_in[1];   // (BK*N, 9)
  const float* rm    = (const float*)d_in[2];   // (N)

  // ---- params in jax tree_flatten (alphabetical) order ----
  int ix = 4;
  auto Fp = [&](int i){ return (const float*)d_in[i]; };
  auto nxt = [&](){ return Fp(ix++); };
  auto getLin = [&](){ Lin l; l.b = nxt(); l.w = nxt(); return l; };
  auto getLN  = [&](){ LNp l; l.b = nxt(); l.g = nxt(); return l; };
  Blk blk[2];
  for (int b = 0; b < 2; ++b) {
    Blk& B = blk[b]; B.has_et = (b == 0);
    if (B.has_et) { B.et_final = getLin(); B.et_init = getLin(); B.et_ln = getLN();
                    B.et_t1 = getLin(); B.et_t2 = getLin(); }
    B.ipa_b = getLin(); B.ipa_hw = nxt(); B.ipa_kv = getLin(); B.ipa_kvp = getLin();
    B.ipa_out = getLin(); B.ipa_q = getLin(); B.ipa_qp = getLin();
    B.ipa_ln = getLN();
    B.nt1 = getLin(); B.nt2 = getLin(); B.nt3 = getLin(); B.nt_ln = getLN();
    B.post = getLin();
    for (int t = 0; t < 2; ++t) { Tf& T = B.tf[t];
      T.b_in = nxt(); T.b_out = nxt(); T.l1 = getLin(); T.l2 = getLin();
      T.ln1 = getLN(); T.ln2 = getLN(); T.w_in = nxt(); T.w_out = nxt(); }
  }
  Lin ep1 = getLin(), ep2 = getLin(), ep3 = getLin(); LNp epln = getLN();
  Lin relposL = getLin(), spL = getLin(), nodeL = getLin();

  // ---- workspace bump allocator ----
  char* base = (char*)d_ws; size_t off = 0;
  auto alloc = [&](size_t bytes){ void* p = base + off; off += (bytes + 255) & ~(size_t)255; return p; };
  auto fb = [&](LL n){ return (float*)alloc((size_t)n * 4); };
  auto hb = [&](LL n){ return (u16*)alloc((size_t)n * 2); };

  u16*  wT       = hb(2432 * 256);
  float* pe_pos  = fb(NSEQ * CPOSC);  u16* pe_pos_bf = hb(NSEQ * CPOSC);
  float* node0   = fb(NSEQ * CS);
  float* p_i     = fb(NSEQ * FEATC);
  float* pe_rel  = fb(NNe * FEATC);   u16* pe_rel_bf = hb(NNe * FEATC);
  float* relpos  = fb(NNe * FEATC);
  u16*  featsE   = hb(BKNN * 214);
  u16*  hE       = hb(BKNN * CZ);     u16* hE2 = hb(BKNN * CZ);
  float* edge    = fb(BKNN * CZ);     float* tmp_edge = fb(BKNN * CZ);
  float* node    = fb(BKN * CS);      u16* node_bf = hb(BKN * CS);
  float* sX      = fb(BKN * CS);      u16* s_bf    = hb(BKN * CS);
  u16*  q_bf     = hb(BKN * 1024);    u16* kv_bf   = hb(BKN * 2048);
  float* qpt     = fb(BKN * 192);     float* kvpt  = fb(BKN * 480);
  float* qp      = fb(BKN * 192);     float* kvp   = fb(BKN * 480);
  float* bproj   = fb(BKNN * 8);
  float* logits  = fb((LL)BKC * 8 * NNe);  u16* a_bf = hb((LL)BKC * 8 * NNe);
  u16*  vT       = hb((LL)BKC * 8 * CHn * NSEQ);
  u16*  vpT      = hb((LL)BKC * 8 * 36 * NSEQ);
  u16*  zT       = hb(BKN * CZ * NSEQ);
  float* o_buf   = fb(BKN * 1024);    float* op_buf = fb(BKN * 288);
  float* opair   = fb(BKN * 1024);    u16* feat_bf  = hb(BKN * 2432);
  float* tmp_a   = fb(BKN * CS);      float* tmp_b  = fb(BKN * CS);
  u16*  ffn_bf   = hb(BKN * CS);      u16* ffn2_bf  = hb(BKN * CS);
  u16*  qkv_bf   = hb(BKN * 768);
  float* mb      = fb(256);
  float* nbuf    = fb(BKN * CZ);
  u16*  ee_bf    = hb(BKNN * 384);    u16* hh1 = hb(BKNN * 384); u16* hh2 = hb(BKNN * 384);

  // ---- launch helpers ----
  auto cvt = [&](const float* src, u16* dst, LL n){
    int g = (int)((n + 255) / 256); if (g > 8192) g = 8192;
    k_cvt_bf16<<<g, 256, 0, stream>>>(src, dst, n);
  };
  auto transA = [&](const void* in, u16* out, int R, int Cc, LL irs, LL ics,
                    LL ibs1, LL ibs2, int binner, int batch, LL obs, int inbf){
    LL tot = (LL)R * Cc; int g = (int)((tot + 255) / 256); if (g > 2048) g = 2048;
    k_transpose_any<<<dim3((unsigned)g, (unsigned)batch), 256, 0, stream>>>(
        in, out, R, Cc, irs, ics, ibs1, ibs2, binner, obs, inbf);
  };
  auto wtr = [&](const float* w, int din, int dout) -> const u16* {
    transA(w, wT, din, dout, dout, 1, 0, 0, 1, 1, (LL)din * dout, 0);
    return wT;
  };
  auto gemm = [&](const u16* A, int lda, const u16* Bm, int ldb, const float* bias,
                  void* C, int ldc, int M, int Ng, int K, float alpha, int relu, int cbf,
                  int batch, int binner, LL sA1, LL sA2, LL sB1, LL sB2, LL sC1, LL sC2){
    GemmP p{A, Bm, bias, C, M, Ng, K, lda, ldb, ldc, sA1, sA2, sB1, sB2, sC1, sC2,
            binner < 1 ? 1 : binner, alpha, relu, cbf};
    dim3 g((Ng + BN - 1) / BN, (M + BM - 1) / BM, batch < 1 ? 1 : batch);
    k_gemm_nt<<<g, 256, 0, stream>>>(p);
  };
  auto gemm1 = [&](const u16* A, int lda, const u16* Bm, int ldb, const float* bias,
                   void* C, int ldc, int M, int Ng, int K, float alpha, int relu, int cbf){
    gemm(A, lda, Bm, ldb, bias, C, ldc, M, Ng, K, alpha, relu, cbf, 1, 1, 0, 0, 0, 0, 0, 0);
  };
  auto lnorm = [&](const float* x, const float* resid, LNp w, float* y, LL rows, int D,
                   const float* rmask, int mask2d){
    k_layernorm<<<(unsigned)rows, 256, 0, stream>>>(x, resid, w.g, w.b, y, D, rmask, NSEQ, mask2d);
  };
  auto softmax = [&](const float* x, const float* addv, u16* out, LL rows){
    k_softmax_bf16<<<(unsigned)rows, 256, 0, stream>>>(x, addv, out, NSEQ);
  };

  // ================= pipeline =================
  k_maskbias<<<1, 256, 0, stream>>>(rm, mb, NSEQ, -1e9f);

  // node embedding
  k_index_embed<<<128, 256, 0, stream>>>(pe_pos, (LL)NSEQ * CPOSC, CPOSC, 0, NSEQ);
  cvt(pe_pos, pe_pos_bf, (LL)NSEQ * CPOSC);
  gemm1(pe_pos_bf, CPOSC, wtr(nodeL.w, CPOSC, CS), CPOSC, nodeL.b, node0, CS, NSEQ, CS, CPOSC, 1.f, 0, 0);
  k_bcast_mask<<<1024, 256, 0, stream>>>(node0, rm, node, BKC, NSEQ, CS);
  cvt(node, node_bf, BKN * CS);
  gemm1(node_bf, CS, wtr(spL.w, CS, FEATC), CS, spL.b, p_i, FEATC, NSEQ, FEATC, CS, 1.f, 0, 0);

  // relpos embedding
  k_index_embed<<<8192, 256, 0, stream>>>(pe_rel, NNe * FEATC, FEATC, 1, NSEQ);
  cvt(pe_rel, pe_rel_bf, NNe * FEATC);
  gemm1(pe_rel_bf, FEATC, wtr(relposL.w, FEATC, FEATC), FEATC, relposL.b, relpos, FEATC,
        (int)NNe, FEATC, FEATC, 1.f, 0, 0);

  // edge features + edge_proj MLP
  k_edge_feats<<<32768, 256, 0, stream>>>(p_i, relpos, trans, featsE, BKC, NSEQ);
  gemm1(featsE, 214, wtr(ep1.w, 214, CZ), 214, ep1.b, hE,  CZ, (int)BKNN, CZ, 214, 1.f, 1, 1);
  gemm1(hE, CZ,     wtr(ep2.w, CZ, CZ),  CZ,  ep2.b, hE2, CZ, (int)BKNN, CZ, CZ,  1.f, 1, 1);
  gemm1(hE2, CZ,    wtr(ep3.w, CZ, CZ),  CZ,  ep3.b, tmp_edge, CZ, (int)BKNN, CZ, CZ, 1.f, 0, 0);
  lnorm(tmp_edge, nullptr, epln, edge, BKNN, CZ, rm, 1);

  for (int b = 0; b < 2; ++b) {
    Blk& B = blk[b];
    // ---- IPA ----
    cvt(node, node_bf, BKN * CS);
    gemm1(node_bf, CS, wtr(B.ipa_q.w,  CS, 1024), CS, B.ipa_q.b,  q_bf,  1024, (int)BKN, 1024, CS, 1.f, 0, 1);
    gemm1(node_bf, CS, wtr(B.ipa_kv.w, CS, 2048), CS, B.ipa_kv.b, kv_bf, 2048, (int)BKN, 2048, CS, 1.f, 0, 1);
    gemm1(node_bf, CS, wtr(B.ipa_qp.w, CS, 192),  CS, B.ipa_qp.b, qpt,   192,  (int)BKN, 192,  CS, 1.f, 0, 0);
    gemm1(node_bf, CS, wtr(B.ipa_kvp.w, CS, 480), CS, B.ipa_kvp.b, kvpt, 480,  (int)BKN, 480,  CS, 1.f, 0, 0);
    k_rot_points<<<512, 256, 0, stream>>>(qpt,  rot, trans, qp,  BKN, 64);
    k_rot_points<<<512, 256, 0, stream>>>(kvpt, rot, trans, kvp, BKN, 160);
    cvt(edge, hE, BKNN * CZ);   // reuse hE as bf16 edge
    gemm1(hE, CZ, wtr(B.ipa_b.w, CZ, HHn), CZ, B.ipa_b.b, bproj, HHn, (int)BKNN, HHn, CZ, 1.f, 0, 0);
    // a = scale * q k^T  (batched over BK*H)
    gemm(q_bf, 1024, kv_bf, 2048, nullptr, logits, NSEQ, NSEQ, NSEQ, CHn,
         0.05103103630798287f, 0, 0, BKC * HHn, HHn,
         (LL)NSEQ * 1024, 128, (LL)NSEQ * 2048, 256, (LL)HHn * NNe, NNe);
    k_pt_bias<<<8192, 256, 0, stream>>>(logits, bproj, qp, kvp, B.ipa_hw, rm, BKC, NSEQ);
    softmax(logits, nullptr, a_bf, (LL)BKC * HHn * NSEQ);
    // o = a @ v
    transA(kv_bf + 128, vT, NSEQ, CHn, 2048, 1, (LL)NSEQ * 2048, 256, HHn, BKC * HHn,
           (LL)CHn * NSEQ, 1);
    gemm(a_bf, NSEQ, vT, NSEQ, nullptr, o_buf, 1024, NSEQ, CHn, NSEQ, 1.f, 0, 0,
         BKC * HHn, HHn, (LL)HHn * NNe, NNe, (LL)HHn * CHn * NSEQ, (LL)CHn * NSEQ,
         (LL)NSEQ * 1024, CHn);
    // op = a @ vp
    transA(kvp + 24, vpT, NSEQ, 36, 480, 1, (LL)NSEQ * 480, 60, HHn, BKC * HHn,
           (LL)36 * NSEQ, 0);
    gemm(a_bf, NSEQ, vpT, NSEQ, nullptr, op_buf, 288, NSEQ, 36, NSEQ, 1.f, 0, 0,
         BKC * HHn, HHn, (LL)HHn * NNe, NNe, (LL)HHn * 36 * NSEQ, (LL)36 * NSEQ,
         (LL)NSEQ * 288, 36);
    // o_pair = a @ z  (batched over (b,i))
    transA(edge, zT, NSEQ, CZ, CZ, 1, NNe * CZ, (LL)NSEQ * CZ, NSEQ, (int)BKN,
           (LL)CZ * NSEQ, 0);
    gemm(a_bf, (int)NNe, zT, NSEQ, nullptr, opair, CZ, HHn, CZ, NSEQ, 1.f, 0, 0,
         (int)BKN, NSEQ, (LL)HHn * NNe, NSEQ, (LL)NSEQ * CZ * NSEQ, (LL)CZ * NSEQ,
         (LL)NSEQ * HHn * CZ, (LL)HHn * CZ);
    // concat feats + out proj + residual LN
    k_gather_feats<<<8192, 256, 0, stream>>>(o_buf, op_buf, opair, rot, trans, feat_bf, BKN);
    gemm1(feat_bf, 2432, wtr(B.ipa_out.w, 2432, CS), 2432, B.ipa_out.b, tmp_a, CS,
          (int)BKN, CS, 2432, 1.f, 0, 0);
    k_maskrow<<<1024, 256, 0, stream>>>(tmp_a, rm, BKN * CS, CS, NSEQ);
    lnorm(tmp_a, node, B.ipa_ln, node, BKN, CS, nullptr, 0);

    // ---- transformer stack on s ----
    (void)hipMemcpyAsync(sX, node, (size_t)BKN * CS * 4, hipMemcpyDeviceToDevice, stream);
    for (int t = 0; t < 2; ++t) {
      Tf& T = B.tf[t];
      cvt(sX, s_bf, BKN * CS);
      gemm1(s_bf, CS, wtr(T.w_in, CS, 768), CS, T.b_in, qkv_bf, 768, (int)BKN, 768, CS, 1.f, 0, 1);
      gemm(qkv_bf, 768, qkv_bf + 256, 768, nullptr, logits, NSEQ, NSEQ, NSEQ, 64,
           0.125f, 0, 0, BKC * 4, 4, (LL)NSEQ * 768, 64, (LL)NSEQ * 768, 64, (LL)4 * NNe, NNe);
      softmax(logits, mb, a_bf, (LL)BKC * 4 * NSEQ);
      transA(qkv_bf + 512, vT, NSEQ, 64, 768, 1, (LL)NSEQ * 768, 64, 4, BKC * 4,
             (LL)64 * NSEQ, 1);
      gemm(a_bf, NSEQ, vT, NSEQ, nullptr, tmp_a, CS, NSEQ, 64, NSEQ, 1.f, 0, 0,
           BKC * 4, 4, (LL)4 * NNe, NNe, (LL)4 * 64 * NSEQ, (LL)64 * NSEQ, (LL)NSEQ * CS, 64);
      cvt(tmp_a, ffn_bf, BKN * CS);
      gemm1(ffn_bf, CS, wtr(T.w_out, CS, CS), CS, T.b_out, tmp_b, CS, (int)BKN, CS, CS, 1.f, 0, 0);
      lnorm(tmp_b, sX, T.ln1, sX, BKN, CS, nullptr, 0);
      cvt(sX, s_bf, BKN * CS);
      gemm1(s_bf, CS, wtr(T.l1.w, CS, CS), CS, T.l1.b, ffn_bf, CS, (int)BKN, CS, CS, 1.f, 1, 1);
      gemm1(ffn_bf, CS, wtr(T.l2.w, CS, CS), CS, T.l2.b, tmp_b, CS, (int)BKN, CS, CS, 1.f, 0, 0);
      lnorm(tmp_b, sX, T.ln2, sX, BKN, CS, nullptr, 0);
    }
    cvt(sX, s_bf, BKN * CS);
    gemm1(s_bf, CS, wtr(B.post.w, CS, CS), CS, B.post.b, tmp_a, CS, (int)BKN, CS, CS, 1.f, 0, 0);
    k_add<<<1024, 256, 0, stream>>>(node, node, tmp_a, BKN * CS);

    // ---- node transition ----
    cvt(node, node_bf, BKN * CS);
    gemm1(node_bf, CS, wtr(B.nt1.w, CS, CS), CS, B.nt1.b, ffn_bf,  CS, (int)BKN, CS, CS, 1.f, 1, 1);
    gemm1(ffn_bf, CS, wtr(B.nt2.w, CS, CS), CS, B.nt2.b, ffn2_bf, CS, (int)BKN, CS, CS, 1.f, 1, 1);
    gemm1(ffn2_bf, CS, wtr(B.nt3.w, CS, CS), CS, B.nt3.b, tmp_a,  CS, (int)BKN, CS, CS, 1.f, 0, 0);
    lnorm(tmp_a, node, B.nt_ln, node, BKN, CS, rm, 0);

    // ---- edge transition ----
    if (B.has_et) {
      cvt(node, node_bf, BKN * CS);
      gemm1(node_bf, CS, wtr(B.et_init.w, CS, CZ), CS, B.et_init.b, nbuf, CZ, (int)BKN, CZ, CS, 1.f, 0, 0);
      k_edge_concat<<<32768, 256, 0, stream>>>(edge, nbuf, (const u16*)nullptr, ee_bf, BKC, NSEQ);
      gemm1(ee_bf, 384, wtr(B.et_t1.w, 384, 384), 384, B.et_t1.b, hh1, 384, (int)BKNN, 384, 384, 1.f, 1, 1);
      gemm1(hh1,  384, wtr(B.et_t2.w, 384, 384), 384, B.et_t2.b, hh2, 384, (int)BKNN, 384, 384, 1.f, 1, 1);
      k_edge_concat<<<32768, 256, 0, stream>>>(edge, nbuf, hh2, hh1, BKC, NSEQ);
      gemm1(hh1, 384, wtr(B.et_final.w, 384, CZ), 384, B.et_final.b, tmp_edge, CZ, (int)BKNN, CZ, 384, 1.f, 0, 0);
      lnorm(tmp_edge, nullptr, B.et_ln, edge, BKNN, CZ, rm, 1);
    }
  }

  (void)hipMemcpyAsync(d_out, node, (size_t)BKN * CS * 4, hipMemcpyDeviceToDevice, stream);
}
